// NodeRelAttn_52106543235746
// MI455X (gfx1250) — matrix operand order repaired and checked by reading, not compiled, not run on hardware
//
#include <hip/hip_runtime.h>

#define NDIM 256
#define EDIM 64
#define HEADS 8
#define DH 32
#define NB 2
#define NN 512
#define TJ 64
#define PST 36   // f32 row stride for projection tiles in LDS (bank-friendly, 16B aligned rows)
#define EST 72   // bf16 row stride for edge tile in LDS

typedef __attribute__((ext_vector_type(16))) __bf16      v16bf;
typedef __attribute__((ext_vector_type(8)))  float        v8f;
typedef __attribute__((ext_vector_type(8)))  unsigned int v8u;

__device__ __forceinline__ unsigned short f2bf(float f) {
  unsigned u = __builtin_bit_cast(unsigned, f);
  unsigned r = u + 0x7fffu + ((u >> 16) & 1u);   // round-to-nearest-even
  return (unsigned short)(r >> 16);
}
__device__ __forceinline__ float bf2f(unsigned short s) {
  unsigned u = ((unsigned)s) << 16;
  return __builtin_bit_cast(float, u);
}
__device__ __forceinline__ unsigned pack2(float a, float b) {
  return (unsigned)f2bf(a) | ((unsigned)f2bf(b) << 16);
}
__device__ __forceinline__ v16bf make_frag(uint4 lo, uint4 hi) {
  v8u r;
  r[0]=lo.x; r[1]=lo.y; r[2]=lo.z; r[3]=lo.w;
  r[4]=hi.x; r[5]=hi.y; r[6]=hi.z; r[7]=hi.w;
  return __builtin_bit_cast(v16bf, r);
}
__device__ __forceinline__ v8f vzero8() {
  v8f c;
  #pragma unroll
  for (int i=0;i<8;i++) c[i] = 0.0f;
  return c;
}
__device__ __forceinline__ v8f wmma_bf16(v16bf a, v16bf b, v8f c) {
  // D = A(16x32 bf16) x B(32x16 bf16) + C(16x16 f32)
  return __builtin_amdgcn_wmma_f32_16x16x32_bf16(false, a, false, b, (short)0, c, false, false);
}
// CDNA5 async global->LDS copy, 16B per lane, GVS addressing (SGPR base + VGPR offset).
// ldsAddr = LDS byte address (low 32 bits of flat shared pointer), tracked by ASYNCcnt.
__device__ __forceinline__ void async_g2l_b128(unsigned ldsAddr, unsigned gOff,
                                               unsigned long long gBase) {
  asm volatile("global_load_async_to_lds_b128 %0, %1, %2 offset:0"
               :: "v"(ldsAddr), "v"(gOff), "s"(gBase) : "memory");
}
__device__ __forceinline__ void wait_asynccnt0() {
  asm volatile("s_wait_asynccnt 0" ::: "memory");
}

// ---------------- f32 -> bf16 conversion (vectorized, 4 elems/thread) ----------------
__global__ void cvt_f32_bf16(const float* __restrict__ src, unsigned short* __restrict__ dst, int n4) {
  int i = blockIdx.x * blockDim.x + threadIdx.x;
  if (i < n4) {
    float4 f = ((const float4*)src)[i];
    uint2 p;
    p.x = pack2(f.x, f.y);
    p.y = pack2(f.z, f.w);
    ((uint2*)dst)[i] = p;
  }
}

// ---------------- node projections: nq (f32, bias bnq+beq), nk/nv (bf16, biases folded) ----
__global__ __launch_bounds__(256) void node_proj(
    const unsigned short* __restrict__ Xbf,
    const unsigned short* __restrict__ Wq, const unsigned short* __restrict__ Wk,
    const unsigned short* __restrict__ Wv,
    const float* __restrict__ bnq, const float* __restrict__ beq,
    const float* __restrict__ bnk, const float* __restrict__ bek,
    const float* __restrict__ bnv, const float* __restrict__ bev,
    float* __restrict__ nqF, unsigned short* __restrict__ nkB, unsigned short* __restrict__ nvB)
{
  const int lane = threadIdx.x & 31;
  const int wave = threadIdx.x >> 5;
  const int rt   = blockIdx.x * 8 + wave;   // row tile 0..63 (16 rows of 1024)
  const int ct   = blockIdx.y;              // col tile 0..15
  const int mat  = blockIdx.z;              // 0=q,1=k,2=v

  const unsigned short* W = (mat==0) ? Wq : (mat==1) ? Wk : Wv;
  const int arow = rt*16 + (lane & 15);
  const int akb  = (lane < 16) ? 0 : 16;    // A K-offset within chunk (contiguous K, matches B)
  const int ncol = ct*16 + (lane & 15);
  const int bko  = (lane < 16) ? 0 : 16;    // B K-offset within chunk (contiguous K, matches A)

  v8f acc = vzero8();
  #pragma unroll
  for (int kc = 0; kc < 8; ++kc) {          // K = 256 in chunks of 32
    const uint4* pa = (const uint4*)(Xbf + (size_t)arow*256 + kc*32 + akb);
    v16bf A = make_frag(pa[0], pa[1]);      // 16 contiguous K kb..kb+15
    const uint4* pb = (const uint4*)(W + (size_t)ncol*256 + kc*32 + bko);
    v16bf Bv = make_frag(pb[0], pb[1]);     // 16 contiguous K of column ncol
    acc = wmma_bf16(A, Bv, acc);
  }

  const float* b1 = (mat==0) ? bnq : (mat==1) ? bnk : bnv;
  const float* b2 = (mat==0) ? beq : (mat==1) ? bek : bev;
  const float bias = b1[ncol] + b2[ncol];
  const int rb = rt*16 + ((lane < 16) ? 0 : 8);
  if (mat == 0) {
    #pragma unroll
    for (int r=0;r<8;r++) nqF[(size_t)(rb+r)*256 + ncol] = acc[r] + bias;
  } else {
    unsigned short* dst = (mat==1) ? nkB : nvB;
    #pragma unroll
    for (int r=0;r<8;r++) dst[(size_t)(rb+r)*256 + ncol] = f2bf(acc[r] + bias);
  }
}

// ---------------- fused edge projection + relational attention --------------------------
// One workgroup per (b,i); wave h handles head h; online softmax over j in tiles of 64.
__global__ __launch_bounds__(256) void edge_attn(
    const float* __restrict__ edge,
    const float* __restrict__ nqF,
    const unsigned short* __restrict__ nkB,
    const unsigned short* __restrict__ nvB,
    const unsigned short* __restrict__ Weq,
    const unsigned short* __restrict__ Wek,
    const unsigned short* __restrict__ Wev,
    float* __restrict__ out)
{
  extern __shared__ __align__(16) unsigned char smem[];
  unsigned short* Ebf   = (unsigned short*)(smem);            // [64][72] bf16   (9216 B)
  unsigned short* nkLds = (unsigned short*)(smem + 9216);     // [64][256] bf16  (32768 B)
  unsigned short* nvLds = (unsigned short*)(smem + 41984);    // [64][256] bf16  (32768 B)
  float* Pbase = (float*)(smem + 74752);                      // [3*8][64*36] f32 (221184 B)
  float* pScr  = (float*)(smem + 295936);                     // [8][64] f32
  float* qv    = (float*)(smem + 297984);                     // [8][32] f32

  const int tid  = threadIdx.x;
  const int lane = tid & 31;
  const int head = tid >> 5;
  const int ii   = blockIdx.x;
  const int bb   = blockIdx.y;

  // LDS byte addresses for the async DMA destinations (low 32 bits of flat shared ptr)
  const unsigned nkLdsAddr = (unsigned)(unsigned long long)(uintptr_t)nkLds;
  const unsigned nvLdsAddr = (unsigned)(unsigned long long)(uintptr_t)nvLds;

  // per-head query offset vector: nq already contains bnq+beq
  qv[head*32 + lane] = nqF[(size_t)(bb*NN + ii)*256 + head*32 + lane];

  // B fragments (per-head edge-weight slices), held in VGPRs for the whole block
  v16bf Bf[3][2][2];
  {
    const unsigned short* Ws[3] = {Weq, Wek, Wev};
    const int bko = (lane < 16) ? 0 : 16;
    #pragma unroll
    for (int m=0;m<3;m++) {
      #pragma unroll
      for (int ct=0;ct<2;ct++) {
        const int col = head*32 + ct*16 + (lane & 15);
        #pragma unroll
        for (int kc=0;kc<2;kc++) {
          const uint4* pb = (const uint4*)(Ws[m] + (size_t)col*64 + kc*32 + bko);
          Bf[m][ct][kc] = make_frag(pb[0], pb[1]);
        }
      }
    }
  }

  float o     = 0.0f;     // lane == d accumulator
  float m_run = -1e30f;
  float l_run = 0.0f;

  const size_t rowBase = ((size_t)(bb*NN + ii)) * NN * EDIM;

  for (int jt = 0; jt < NN/TJ; ++jt) {
    const int j0 = jt * TJ;
    __syncthreads();

    // -- kick off nk/nv tile staging with CDNA5 async global->LDS DMA (ASYNCcnt path)
    {
      const unsigned long long snk =
          (unsigned long long)(uintptr_t)(nkB + (size_t)(bb*NN + j0)*256);
      const unsigned long long snv =
          (unsigned long long)(uintptr_t)(nvB + (size_t)(bb*NN + j0)*256);
      #pragma unroll
      for (int k=0;k<8;k++) {
        const unsigned goff = (unsigned)((tid + k*256) * 16);   // 2048 x 16B = 32KB
        async_g2l_b128(nkLdsAddr + goff, goff, snk);
        async_g2l_b128(nvLdsAddr + goff, goff, snv);
      }
    }

    // -- stage edge tile [64][64] f32 -> bf16 LDS (contiguous 16KB read + convert)
    {
      const int j  = tid >> 2;
      const int c0 = (tid & 3) * 16;
      const float4* src = (const float4*)(edge + rowBase + (size_t)(j0 + j)*EDIM + c0);
      float4 f0 = src[0], f1 = src[1], f2 = src[2], f3 = src[3];
      uint4 d0, d1;
      d0.x = pack2(f0.x, f0.y); d0.y = pack2(f0.z, f0.w);
      d0.z = pack2(f1.x, f1.y); d0.w = pack2(f1.z, f1.w);
      d1.x = pack2(f2.x, f2.y); d1.y = pack2(f2.z, f2.w);
      d1.z = pack2(f3.x, f3.y); d1.w = pack2(f3.z, f3.w);
      uint4* dst = (uint4*)(Ebf + j*EST + c0);
      dst[0] = d0; dst[1] = d1;
    }
    // -- prefetch next edge tile (global_prefetch_b8)
    if (jt < NN/TJ - 1) {
      const char* pf = (const char*)(edge + rowBase + (size_t)(j0 + TJ)*EDIM) + tid*64;
      __builtin_prefetch(pf, 0, 1);
    }
    wait_asynccnt0();      // async LDS writes from this wave complete
    __syncthreads();       // ...and visible to all waves

    // -- per-head edge projections: Pq/Pk/Pv[64x32] = Etile[64x64] @ W_h^T[64x32]
    {
      const int am  = lane & 15;
      const int akb = (lane < 16) ? 0 : 16;
      const int col = lane & 15;
      const int rbo = (lane < 16) ? 0 : 8;
      #pragma unroll
      for (int rt=0; rt<4; ++rt) {
        const int arow = rt*16 + am;
        const uint4* pa0 = (const uint4*)(Ebf + arow*EST + akb);
        const uint4* pa1 = (const uint4*)(Ebf + arow*EST + 32 + akb);
        v16bf A0 = make_frag(pa0[0], pa0[1]);
        v16bf A1 = make_frag(pa1[0], pa1[1]);
        #pragma unroll
        for (int m=0;m<3;m++) {
          float* Pm = Pbase + (size_t)(m*8 + head) * (TJ*PST);
          #pragma unroll
          for (int ct=0;ct<2;ct++) {
            v8f c = vzero8();
            c = wmma_bf16(A0, Bf[m][ct][0], c);
            c = wmma_bf16(A1, Bf[m][ct][1], c);
            const int cc = ct*16 + col;
            const int rb = rt*16 + rbo;
            #pragma unroll
            for (int r=0;r<8;r++) Pm[(rb + r)*PST + cc] = c[r];
          }
        }
      }
    }
    __syncthreads();

    // -- scores s_j = (Pq_j + qv) . (Pk_j + nk_j), 2 j's per lane
    const float* Pq  = Pbase + (size_t)(0*8 + head) * (TJ*PST);
    const float* Pk  = Pbase + (size_t)(1*8 + head) * (TJ*PST);
    const float* Pv  = Pbase + (size_t)(2*8 + head) * (TJ*PST);
    const float* qvp = qv + head*32;

    float s[2];
    #pragma unroll
    for (int t=0;t<2;t++) {
      const int j = lane + t*32;
      const float* pq = Pq + j*PST;
      const float* pk = Pk + j*PST;
      const uint4* nk4 = (const uint4*)(nkLds + j*256 + head*32);
      float acc = 0.0f;
      #pragma unroll
      for (int dd=0; dd<4; ++dd) {
        uint4 nkv = nk4[dd];
        unsigned nw[4] = {nkv.x, nkv.y, nkv.z, nkv.w};
        #pragma unroll
        for (int e=0;e<4;e++) {
          const int d = dd*8 + e*2;
          const float k0 = pk[d]   + bf2f((unsigned short)(nw[e] & 0xffffu));
          const float k1 = pk[d+1] + bf2f((unsigned short)(nw[e] >> 16));
          acc += (pq[d]   + qvp[d]  ) * k0;
          acc += (pq[d+1] + qvp[d+1]) * k1;
        }
      }
      s[t] = acc * 0.17677669529663687f;   // 1/sqrt(32)
    }

    // -- online softmax (wave32 butterfly reductions)
    float mt = fmaxf(s[0], s[1]);
    #pragma unroll
    for (int off=16; off>=1; off>>=1) mt = fmaxf(mt, __shfl_xor(mt, off, 32));
    const float mnew  = fmaxf(m_run, mt);
    const float alpha = __expf(m_run - mnew);
    const float p0 = __expf(s[0] - mnew);
    const float p1 = __expf(s[1] - mnew);
    float ps = p0 + p1;
    #pragma unroll
    for (int off=16; off>=1; off>>=1) ps += __shfl_xor(ps, off, 32);
    l_run = l_run * alpha + ps;
    m_run = mnew;
    pScr[head*64 + lane]      = p0;   // same-wave LDS: in-order, no barrier needed
    pScr[head*64 + lane + 32] = p1;
    o *= alpha;

    // -- PV accumulation: lane == d
    const int d = lane;
    #pragma unroll 4
    for (int j=0;j<TJ;j++) {
      const float p = pScr[head*64 + j];
      const float v = Pv[j*PST + d] + bf2f(nvLds[j*256 + head*32 + d]);
      o += p * v;
    }
  }

  out[(size_t)(bb*NN + ii)*256 + head*32 + lane] = o / l_run;
}

// ---------------------------------------------------------------------------------------
extern "C" void kernel_launch(void* const* d_in, const int* in_sizes, int n_in,
                              void* d_out, int out_size, void* d_ws, size_t ws_size,
                              hipStream_t stream) {
  (void)in_sizes; (void)n_in; (void)out_size; (void)ws_size;
  const float* node = (const float*)d_in[0];
  const float* edge = (const float*)d_in[1];
  const float* Wnq  = (const float*)d_in[2];  const float* bnq = (const float*)d_in[3];
  const float* Wnk  = (const float*)d_in[4];  const float* bnk = (const float*)d_in[5];
  const float* Wnv  = (const float*)d_in[6];  const float* bnv = (const float*)d_in[7];
  const float* Weq  = (const float*)d_in[8];  const float* beq = (const float*)d_in[9];
  const float* Wek  = (const float*)d_in[10]; const float* bek = (const float*)d_in[11];
  const float* Wev  = (const float*)d_in[12]; const float* bev = (const float*)d_in[13];
  float* out = (float*)d_out;

  unsigned char* w = (unsigned char*)d_ws;   // ~3.0 MB used
  unsigned short* Xbf  = (unsigned short*)(w + 0);        // [1024][256] bf16
  unsigned short* WnqB = (unsigned short*)(w + 524288);   // [256][256] bf16
  unsigned short* WnkB = (unsigned short*)(w + 655360);
  unsigned short* WnvB = (unsigned short*)(w + 786432);
  unsigned short* WeqB = (unsigned short*)(w + 917504);   // [256][64] bf16
  unsigned short* WekB = (unsigned short*)(w + 950272);
  unsigned short* WevB = (unsigned short*)(w + 983040);
  float*          nqF  = (float*)         (w + 1015808);  // [1024][256] f32 (+bnq+beq)
  unsigned short* nkB  = (unsigned short*)(w + 2064384);  // [1024][256] bf16 (+bnk+bek)
  unsigned short* nvB  = (unsigned short*)(w + 2588672);  // [1024][256] bf16 (+bnv+bev)

  auto cvt = [&](const float* s, unsigned short* d, int n) {
    int n4 = n / 4;
    cvt_f32_bf16<<<(n4 + 255)/256, 256, 0, stream>>>(s, d, n4);
  };
  cvt(node, Xbf, NB*NN*NDIM);
  cvt(Wnq, WnqB, NDIM*NDIM); cvt(Wnk, WnkB, NDIM*NDIM); cvt(Wnv, WnvB, NDIM*NDIM);
  cvt(Weq, WeqB, NDIM*EDIM); cvt(Wek, WekB, NDIM*EDIM); cvt(Wev, WevB, NDIM*EDIM);

  node_proj<<<dim3(8,16,3), 256, 0, stream>>>(Xbf, WnqB, WnkB, WnvB,
      bnq, beq, bnk, bek, bnv, bev, nqF, nkB, nvB);

  edge_attn<<<dim3(NN, NB), 256, 299008, stream>>>(edge, nqF, nkB, nvB,
      WeqB, WekB, WevB, out);
}
